// CausalSelfAttention_8632884265122
// MI455X (gfx1250) — matrix operand order repaired and checked by reading, not compiled, not run on hardware
//
#include <hip/hip_runtime.h>

typedef __attribute__((ext_vector_type(16))) __bf16 v16bf;
typedef __attribute__((ext_vector_type(8)))  __bf16 v8bf;
typedef __attribute__((ext_vector_type(8)))  float  v8f;

union Frag { v16bf v; v8bf h[2]; };

constexpr int BB  = 2;
constexpr int T   = 2048;
constexpr int C   = 768;
constexpr int H   = 12;
constexpr int Dh  = 64;
constexpr int N3C = 3 * C;    // 2304
constexpr int M   = BB * T;   // 4096

__device__ __forceinline__ __bf16 f2bf(float f) {
  unsigned int u = __builtin_bit_cast(unsigned int, f);
  u += 0x7FFFu + ((u >> 16) & 1u);                 // round-to-nearest-even
  return __builtin_bit_cast(__bf16, (unsigned short)(u >> 16));
}

__device__ __forceinline__ v8f wmma_bf16(const Frag& a, const Frag& b, v8f c) {
  return __builtin_amdgcn_wmma_f32_16x16x32_bf16(false, a.v, false, b.v,
                                                 (short)0, c, false, false);
}

// XOR lane-exchange via DS_SWIZZLE_B32 (group-of-32 mode, immediate pattern):
// offset[4:0]=and_mask(0x1F), offset[9:5]=or_mask(0), offset[14:10]=xor_mask.
template <int MASK>
__device__ __forceinline__ float swz_xor(float x) {
  int i = __builtin_bit_cast(int, x);
  i = __builtin_amdgcn_ds_swizzle(i, (MASK << 10) | 0x1F);
  return __builtin_bit_cast(float, i);
}
// reduce across the 16 lanes of each half-wave; result broadcast to those lanes
__device__ __forceinline__ float half_max(float x) {
  x = fmaxf(x, swz_xor<8>(x));
  x = fmaxf(x, swz_xor<4>(x));
  x = fmaxf(x, swz_xor<2>(x));
  x = fmaxf(x, swz_xor<1>(x));
  return x;
}
__device__ __forceinline__ float half_sum(float x) {
  x += swz_xor<8>(x);
  x += swz_xor<4>(x);
  x += swz_xor<2>(x);
  x += swz_xor<1>(x);
  return x;
}

// A-fragment (16x32): lane holds row (lane&15); two 8-bf16 runs at K offsets half*8, 16+half*8
__device__ __forceinline__ Frag load_af(const __bf16* row, int kk, int half) {
  Frag a;
  a.h[0] = *(const v8bf*)(row + kk + half * 8);
  a.h[1] = *(const v8bf*)(row + kk + 16 + half * 8);
  return a;
}
// B-fragment (32x16) from transposed storage: lane holds col (lane&15); two 8-bf16 runs at K offsets half*8, 16+half*8
__device__ __forceinline__ Frag load_bf(const __bf16* row, int kk, int half) {
  Frag b;
  b.h[0] = *(const v8bf*)(row + kk + half * 8);
  b.h[1] = *(const v8bf*)(row + kk + 16 + half * 8);
  return b;
}

// ---------------- elementwise fp32 -> bf16 ----------------
__global__ void cast_f32_bf16(const float* __restrict__ in, __bf16* __restrict__ out, int n) {
  int i = blockIdx.x * blockDim.x + threadIdx.x;
  if (i < n) out[i] = f2bf(in[i]);
}

// out[n][k] = bf16(in[k][n]); in is [Kd][Nd] row-major, out is [Nd][Kd]
__global__ void cast_transpose(const float* __restrict__ in, __bf16* __restrict__ out,
                               int Kd, int Nd) {
  int i = blockIdx.x * blockDim.x + threadIdx.x;
  if (i < Kd * Nd) {
    int n = i / Kd, k = i % Kd;
    out[i] = f2bf(in[(size_t)k * Nd + n]);
  }
}

// ---------------- QKV GEMM: 32x64 tile per wave, scattered to Q/K/Vt ----------------
__global__ void __launch_bounds__(256) qkv_wmma(
    const __bf16* __restrict__ xb,    // [M][C]
    const __bf16* __restrict__ wt,    // [N3C][C]  (W_attn transposed)
    const float*  __restrict__ bias,  // [N3C]
    __bf16* __restrict__ Qo,          // [BB,H,T,Dh]
    __bf16* __restrict__ Ko,          // [BB,H,T,Dh]
    __bf16* __restrict__ Vt)          // [BB,H,Dh,T]
{
  const int lane = threadIdx.x & 31;
  const int wave = blockIdx.x * 8 + (threadIdx.x >> 5);
  const int tiles_n = N3C / 64;                 // 36
  const int tm = wave / tiles_n;                // 0..127 (32-row tiles)
  const int tn = wave % tiles_n;                // 0..35  (64-col tiles)
  const int half = lane >> 4, l15 = lane & 15;

  const __bf16* arow0 = xb + (size_t)(tm * 32 + l15) * C;
  const __bf16* arow1 = arow0 + (size_t)16 * C;
  const __bf16* brow[4];
#pragma unroll
  for (int j = 0; j < 4; ++j)
    brow[j] = wt + (size_t)(tn * 64 + j * 16 + l15) * C;

  v8f acc[2][4] = {};
  for (int kk = 0; kk < C; kk += 32) {
    Frag a0 = load_af(arow0, kk, half);
    Frag a1 = load_af(arow1, kk, half);
#pragma unroll
    for (int j = 0; j < 4; ++j) {
      Frag b = load_bf(brow[j], kk, half);
      acc[0][j] = wmma_bf16(a0, b, acc[0][j]);
      acc[1][j] = wmma_bf16(a1, b, acc[1][j]);
    }
  }

  const int which = tn / 12;           // 0=Q, 1=K, 2=V
  const int h = tn % 12;
#pragma unroll
  for (int j = 0; j < 4; ++j) {
    const int n = tn * 64 + j * 16 + l15;
    const int d = j * 16 + l15;
    const float bn = bias[n];
#pragma unroll
    for (int i = 0; i < 2; ++i) {
      const int m0 = tm * 32 + i * 16 + half * 8;
#pragma unroll
      for (int r = 0; r < 8; ++r) {
        const int m = m0 + r;
        const int bi = m / T, t = m % T;
        const __bf16 val = f2bf(acc[i][j][r] + bn);
        const size_t bh = (size_t)bi * H + h;
        if (which == 0)      Qo[(bh * T + t) * Dh + d] = val;
        else if (which == 1) Ko[(bh * T + t) * Dh + d] = val;
        else                 Vt[(bh * Dh + d) * T + t] = val;
      }
    }
  }
}

// ---------------- flash attention: one wave per (b,h,32-query tile) ----------------
__global__ void __launch_bounds__(256) attn_wmma(
    const __bf16* __restrict__ Qo, const __bf16* __restrict__ Ko,
    const __bf16* __restrict__ Vt, __bf16* __restrict__ Y)  // Y: [M][C] bf16
{
  __shared__ __bf16 pbuf[8][2][16][32];   // per-wave P staging tiles
  const int lane = threadIdx.x & 31;
  const int w = threadIdx.x >> 5;
  const int wave = blockIdx.x * 8 + w;
  const int qt32 = T / 32;             // 64
  const int qi = wave % qt32;
  const int bh = wave / qt32;          // b*H + h
  const int half = lane >> 4, l15 = lane & 15;

  const __bf16* Qbase = Qo + (size_t)bh * T * Dh;
  const __bf16* Kbase = Ko + (size_t)bh * T * Dh;
  const __bf16* Vbase = Vt + (size_t)bh * Dh * T;

  Frag qf[2][2];
#pragma unroll
  for (int i = 0; i < 2; ++i) {
    const __bf16* qrow = Qbase + (size_t)(qi * 32 + i * 16 + l15) * Dh;
#pragma unroll
    for (int ds = 0; ds < 2; ++ds) qf[i][ds] = load_af(qrow, ds * 32, half);
  }

  v8f o[2][4] = {};
  float mrow[2][8], lrow[2][8];
#pragma unroll
  for (int i = 0; i < 2; ++i)
#pragma unroll
    for (int r = 0; r < 8; ++r) { mrow[i][r] = -1e30f; lrow[i][r] = 0.f; }

  for (int s = 0; s <= qi; ++s) {
    const int key0 = s * 32;
    // prefetch next step's K/V rows (one line per lane covers the tiles)
    if (s < qi) {
      const int kn = key0 + 32;
      __builtin_prefetch(Kbase + (size_t)(kn + l15) * Dh, 0, 1);
      __builtin_prefetch(Kbase + (size_t)(kn + 16 + l15) * Dh, 0, 1);
      __builtin_prefetch(Vbase + (size_t)(lane * 2) * T + kn, 0, 1);
    }
    // K fragments (shared by both query row-tiles)
    Frag kf[2][2];
#pragma unroll
    for (int sub = 0; sub < 2; ++sub) {
      const __bf16* krow = Kbase + (size_t)(key0 + sub * 16 + l15) * Dh;
#pragma unroll
      for (int ds = 0; ds < 2; ++ds) kf[sub][ds] = load_bf(krow, ds * 32, half);
    }
    // V fragments (shared by both query row-tiles)
    Frag vf[4];
#pragma unroll
    for (int dt = 0; dt < 4; ++dt) {
      const __bf16* vrow = Vbase + (size_t)(dt * 16 + l15) * T + key0;
      vf[dt] = load_bf(vrow, 0, half);
    }
#pragma unroll
    for (int i = 0; i < 2; ++i) {
      v8f sf0 = {}, sf1 = {};
#pragma unroll
      for (int ds = 0; ds < 2; ++ds) {
        sf0 = wmma_bf16(qf[i][ds], kf[0][ds], sf0);
        sf1 = wmma_bf16(qf[i][ds], kf[1][ds], sf1);
      }
      float p0[8], p1[8];
#pragma unroll
      for (int r = 0; r < 8; ++r) {
        const int q = qi * 32 + i * 16 + half * 8 + r;
        float s0 = sf0[r] * 0.125f;              // 1/sqrt(64)
        float s1 = sf1[r] * 0.125f;
        if (key0 + l15 > q)      s0 = -1e30f;    // causal mask
        if (key0 + 16 + l15 > q) s1 = -1e30f;
        const float mx = half_max(fmaxf(s0, s1));
        const float mnew = fmaxf(mrow[i][r], mx);
        const float corr = __expf(mrow[i][r] - mnew);
        mrow[i][r] = mnew;
        p0[r] = __expf(s0 - mnew);
        p1[r] = __expf(s1 - mnew);
        lrow[i][r] = lrow[i][r] * corr + half_sum(p0[r] + p1[r]);
#pragma unroll
        for (int dt = 0; dt < 4; ++dt) o[i][dt][r] *= corr;
        pbuf[w][i][half * 8 + r][l15]      = f2bf(p0[r]);   // C-layout -> LDS
        pbuf[w][i][half * 8 + r][16 + l15] = f2bf(p1[r]);
      }
      Frag pf;                                              // LDS -> A-layout
      pf.h[0] = *(const v8bf*)(&pbuf[w][i][l15][half * 8]);
      pf.h[1] = *(const v8bf*)(&pbuf[w][i][l15][16 + half * 8]);
#pragma unroll
      for (int dt = 0; dt < 4; ++dt)
        o[i][dt] = wmma_bf16(pf, vf[dt], o[i][dt]);
    }
  }

  const int b = bh / H, h = bh % H;
#pragma unroll
  for (int i = 0; i < 2; ++i) {
#pragma unroll
    for (int r = 0; r < 8; ++r) {
      const float inv = 1.0f / lrow[i][r];
      const size_t row = (size_t)b * T + qi * 32 + i * 16 + half * 8 + r;
      __bf16* yr = Y + row * C + h * Dh;
#pragma unroll
      for (int dt = 0; dt < 4; ++dt)
        yr[dt * 16 + l15] = f2bf(o[i][dt][r] * inv);
    }
  }
}

// ---------------- output projection: 32x64 tile per wave, fp32 out ----------------
__global__ void __launch_bounds__(256) proj_wmma(
    const __bf16* __restrict__ yb,    // [M][C]
    const __bf16* __restrict__ wt,    // [C][C] transposed
    const float* __restrict__ bias,   // [C]
    float* __restrict__ out)          // [M][C]
{
  const int lane = threadIdx.x & 31;
  const int wave = blockIdx.x * 8 + (threadIdx.x >> 5);
  const int tiles_n = C / 64;                    // 12
  const int tm = wave / tiles_n, tn = wave % tiles_n;
  const int half = lane >> 4, l15 = lane & 15;

  const __bf16* arow0 = yb + (size_t)(tm * 32 + l15) * C;
  const __bf16* arow1 = arow0 + (size_t)16 * C;
  const __bf16* brow[4];
#pragma unroll
  for (int j = 0; j < 4; ++j)
    brow[j] = wt + (size_t)(tn * 64 + j * 16 + l15) * C;

  v8f acc[2][4] = {};
  for (int kk = 0; kk < C; kk += 32) {
    Frag a0 = load_af(arow0, kk, half);
    Frag a1 = load_af(arow1, kk, half);
#pragma unroll
    for (int j = 0; j < 4; ++j) {
      Frag b = load_bf(brow[j], kk, half);
      acc[0][j] = wmma_bf16(a0, b, acc[0][j]);
      acc[1][j] = wmma_bf16(a1, b, acc[1][j]);
    }
  }
#pragma unroll
  for (int j = 0; j < 4; ++j) {
    const int n = tn * 64 + j * 16 + l15;
    const float bn = bias[n];
#pragma unroll
    for (int i = 0; i < 2; ++i) {
      const int m0 = tm * 32 + i * 16 + half * 8;
#pragma unroll
      for (int r = 0; r < 8; ++r)
        out[(size_t)(m0 + r) * C + n] = acc[i][j][r] + bn;
    }
  }
}

extern "C" void kernel_launch(void* const* d_in, const int* in_sizes, int n_in,
                              void* d_out, int out_size, void* d_ws, size_t ws_size,
                              hipStream_t stream) {
  const float* x      = (const float*)d_in[0];
  const float* W_attn = (const float*)d_in[1];
  const float* b_attn = (const float*)d_in[2];
  const float* W_proj = (const float*)d_in[3];
  const float* b_proj = (const float*)d_in[4];
  float* out = (float*)d_out;

  char* p = (char*)d_ws;
  auto carve = [&](size_t bytes) -> void* {
    void* r = (void*)p;
    p += (bytes + 255) & ~(size_t)255;
    return r;
  };
  __bf16* xb  = (__bf16*)carve((size_t)M * C * 2);
  __bf16* wat = (__bf16*)carve((size_t)N3C * C * 2);
  __bf16* wpt = (__bf16*)carve((size_t)C * C * 2);
  __bf16* Qb  = (__bf16*)carve((size_t)BB * H * T * Dh * 2);
  __bf16* Kb  = (__bf16*)carve((size_t)BB * H * T * Dh * 2);
  __bf16* Vtb = (__bf16*)carve((size_t)BB * H * Dh * T * 2);
  __bf16* Yb  = (__bf16*)carve((size_t)M * C * 2);

  cast_f32_bf16<<<(M * C) / 256, 256, 0, stream>>>(x, xb, M * C);
  cast_transpose<<<(N3C * C) / 256, 256, 0, stream>>>(W_attn, wat, C, N3C);
  cast_transpose<<<(C * C) / 256, 256, 0, stream>>>(W_proj, wpt, C, C);

  qkv_wmma<<<(M / 32) * (N3C / 64) / 8, 256, 0, stream>>>(xb, wat, b_attn, Qb, Kb, Vtb);
  attn_wmma<<<(BB * H * (T / 32)) / 8, 256, 0, stream>>>(Qb, Kb, Vtb, Yb);
  proj_wmma<<<(M / 32) * (C / 64) / 8, 256, 0, stream>>>(Yb, wpt, b_proj, out);
}
